// GeodesicButterflyLayer_49314814492786
// MI455X (gfx1250) — matrix operand order repaired and checked by reading, not compiled, not run on hardware
//
#include <hip/hip_runtime.h>
#include <math.h>

#define DIM      1024
#define LSTAGES  10
#define NPARAMS  2046
#define FEPS     1e-7f
#define MAXT     (1.0f - 1e-5f)
#define TPAR     0.5f

typedef __attribute__((ext_vector_type(16))) __bf16 v16bf;
typedef __attribute__((ext_vector_type(8)))  float  v8f;

// --- CDNA5 async global->LDS path (probe via __has_builtin; fallback = sync copy)
#if defined(__has_builtin)
#if __has_builtin(__builtin_amdgcn_global_load_async_to_lds_b128) && \
    __has_builtin(__builtin_amdgcn_s_wait_asynccnt)
#define HAVE_ASYNC_LDS 1
#endif
#endif
#ifndef HAVE_ASYNC_LDS
#define HAVE_ASYNC_LDS 0
#endif

#if HAVE_ASYNC_LDS
// builtin signature (from hipcc diagnostic): (v4i addrspace(1)*, v4i addrspace(3)*, Imm, Imm)
typedef int v4i_t __attribute__((vector_size(16)));
typedef __attribute__((address_space(1))) v4i_t* as1_v4i;
typedef __attribute__((address_space(3))) v4i_t* as3_v4i;
// generic LDS pointer's low 32 bits are the LDS byte offset; generic global == AS1
#define ASYNC_CP16(gsrc, ldst)                                                  \
  __builtin_amdgcn_global_load_async_to_lds_b128(                               \
      (as1_v4i)(uintptr_t)(gsrc), (as3_v4i)(unsigned)(uintptr_t)(ldst), 0, 0)
#define WAIT_ASYNC(n) __builtin_amdgcn_s_wait_asynccnt(n)
#else
#define WAIT_ASYNC(n) ((void)0)
#endif

// f32 -> bf16 bits, round-to-nearest-even (storage as ushort; no __bf16 math)
__device__ __forceinline__ unsigned short f2bf(float f) {
  unsigned u = __float_as_uint(f);
  u = u + 0x7fffu + ((u >> 16) & 1u);
  return (unsigned short)(u >> 16);
}

__device__ __forceinline__ float atanh_clamped(float n) {
  float a = fminf(n, MAXT);
  return 0.5f * __logf((1.f + a) / (1.f - a));
}

// ---------------------------------------------------------------------------
// Pass 1: build butterfly matrix T (v = u @ T), packed in WMMA-B bf16 layout.
// Packed element (k, n): kt=k>>5, kin=k&31, lane=(kin>>4)*16+(n&15),
// chunk=(kin>>3)&1, t=kin&7, idx = ((((kt*64 + n>>4)*2 + chunk)*32 + lane)*8 + t)
// ---------------------------------------------------------------------------
__global__ __launch_bounds__(256) void k_build_T(const float* __restrict__ params,
                                                 unsigned short* __restrict__ Tpk) {
  __shared__ float vec[DIM];
  __shared__ float par[NPARAMS];
  const int tid = threadIdx.x;
  const int k   = blockIdx.x;
  for (int i = tid; i < NPARAMS; i += 256) par[i] = params[i];
  for (int i = tid; i < DIM; i += 256)     vec[i] = (i == k) ? 1.f : 0.f;
  __syncthreads();
  int off = 0;
  for (int l = 0; l < LSTAGES; ++l) {
    const int bs = 1 << l;
    for (int p = tid; p < DIM / 2; p += 256) {
      int blk = p >> l;
      int pos = p & (bs - 1);
      int i1  = blk * 2 * bs + pos;
      int i2  = i1 + bs;
      float a = par[off + blk * 2];
      float b = par[off + blk * 2 + 1];
      float x1 = vec[i1], x2 = vec[i2];
      vec[i1] =  a * x1 + b * x2;
      vec[i2] = -b * x1 + a * x2;
    }
    off += ((DIM / 2) >> l) * 2;
    __syncthreads();
  }
  const int kt = k >> 5, kin = k & 31;
  const int chunk = (kin >> 3) & 1, t = kin & 7;
  const int lhalf = (kin >> 4) * 16;
  for (int n = tid; n < DIM; n += 256) {
    int ntl = n >> 4;
    int L   = lhalf + (n & 15);
    int idx = (((kt * 64 + ntl) * 2 + chunk) * 32 + L) * 8 + t;
    Tpk[idx] = f2bf(vec[n]);
  }
}

// ---------------------------------------------------------------------------
// Pass 2: per-row h2 = sum(h^2) and log_map scale  (wave32 per row)
// ---------------------------------------------------------------------------
__global__ __launch_bounds__(256) void k_scales(const float* __restrict__ h,
                                                float* __restrict__ s_log,
                                                float* __restrict__ h2out) {
  const int row  = blockIdx.x * 8 + (threadIdx.x >> 5);
  const int lane = threadIdx.x & 31;
  const float* hr = h + (size_t)row * DIM;
  float s = 0.f;
  for (int i = lane; i < DIM; i += 32) { float v = hr[i]; s += v * v; }
  for (int o = 16; o > 0; o >>= 1) s += __shfl_xor(s, o, 32);
  if (lane == 0) {
    float nh = sqrtf(s + FEPS);
    s_log[row] = atanh_clamped(nh) / nh;
    h2out[row] = s;
  }
}

// ---------------------------------------------------------------------------
// Pass 3: fused  v = (s_log * h) @ T  via v_wmma_f32_16x16x32_bf16, with
// exp_map + geodesic epilogue (row dots via LDS ds_add_f32; z = P*h + Q*v).
// Workgroup: 64 rows x 1024 cols; 1024 threads = 32 waves = 4 msub x 8 ngrp.
// Double-buffered pipeline, T slabs fetched with async global->LDS DMA:
//   wait(slab ks) -> barrier -> 8xWMMA(slab ks) -> barrier -> issue slab ks+2
// ---------------------------------------------------------------------------
__global__ __launch_bounds__(1024) void k_gemm(const float* __restrict__ h,
                                               const unsigned short* __restrict__ Tpk,
                                               const float* __restrict__ s_log,
                                               const float* __restrict__ h2arr,
                                               float* __restrict__ out,
                                               int* __restrict__ gflag) {
  extern __shared__ __align__(16) char smem[];
  unsigned short* smT0 = (unsigned short*)smem;             // 64 KB slab buf 0
  unsigned short* smT1 = (unsigned short*)(smem + 65536);   // 64 KB slab buf 1
  unsigned short* smA0 = (unsigned short*)(smem + 131072);  // 64 x 40 bf16 (pad)
  unsigned short* smA1 = (unsigned short*)(smem + 136192);
  float* v2r = (float*)(smem + 141312);                     // [64]
  float* hvr = (float*)(smem + 141568);                     // [64]
  float* Pr  = (float*)(smem + 141824);                     // [64]
  float* Qr  = (float*)(smem + 142080);                     // [64]
  int*   sfl = (int*)(smem + 142336);

  const int tid  = threadIdx.x;
  const int lane = tid & 31;
  const int wv   = tid >> 5;
  const int msub = wv >> 3;        // 0..3 : 16-row sub-tile
  const int ngrp = wv & 7;         // 0..7 : 128-col group
  const int mrow = lane & 15;
  const int hh   = lane >> 4;
  const int row0 = blockIdx.x * 64;

  v8f acc[8] = {};

  // cooperative A-stage assignment: 2 consecutive k-elements per thread
  const int ar = (tid * 2) >> 5;   // row 0..63
  const int ak = (tid * 2) & 31;   // k within slab (even)
  const float sA = s_log[row0 + ar];
  const float* hrowA = h + (size_t)(row0 + ar) * DIM;

  // --- stage helpers -------------------------------------------------------
  auto stageT = [&](int slab, unsigned short* buf) {
#if HAVE_ASYNC_LDS
    #pragma unroll
    for (int i = 0; i < 4; ++i) {
      size_t e = (size_t)(tid + i * 1024) * 8;   // ushort index of 16B chunk
      ASYNC_CP16(Tpk + (size_t)slab * 32768 + e, buf + e);
    }
#else
    __builtin_prefetch(Tpk + (size_t)slab * 32768 + tid * 32, 0, 1);
    const float4* s = (const float4*)(Tpk + (size_t)slab * 32768);
    float4* d = (float4*)buf;
    #pragma unroll
    for (int i = 0; i < 4; ++i) d[tid + i * 1024] = s[tid + i * 1024];
#endif
  };
  auto stageA = [&](int slab, unsigned short* buf) {
    float2 hv2 = *(const float2*)(hrowA + slab * 32 + ak);
    buf[ar * 40 + ak]     = f2bf(hv2.x * sA);
    buf[ar * 40 + ak + 1] = f2bf(hv2.y * sA);
  };

  // prologue: slabs 0 and 1 in flight
  stageT(0, smT0); stageA(0, smA0);
  stageT(1, smT1); stageA(1, smA1);

  for (int ks = 0; ks < 32; ++ks) {
    unsigned short* tcur = (ks & 1) ? smT1 : smT0;
    unsigned short* acur = (ks & 1) ? smA1 : smA0;

    if (ks < 31) WAIT_ASYNC(4);    // retire slab ks, leave ks+1 in flight
    else         WAIT_ASYNC(0);
    __syncthreads();               // slab ks visible to all waves

    // A fragment: lane m = mrow, K-range {hh*16..+15} contiguous
    v16bf afrag;
    {
      int eo = (msub * 16 + mrow) * 40 + hh * 16;
      *(float4*)&afrag       = *(const float4*)(acur + eo);
      *(((float4*)&afrag)+1) = *(const float4*)(acur + eo + 8);
    }
    #pragma unroll
    for (int nt = 0; nt < 8; ++nt) {
      int ntl = ngrp * 8 + nt;
      v16bf bfrag;  // lane n = mrow(+half), 16 contiguous K values per lane
      *(float4*)&bfrag       = *(const float4*)(tcur + ((ntl * 2 + 0) * 32 + lane) * 8);
      *(((float4*)&bfrag)+1) = *(const float4*)(tcur + ((ntl * 2 + 1) * 32 + lane) * 8);
      acc[nt] = __builtin_amdgcn_wmma_f32_16x16x32_bf16(
          false, afrag, false, bfrag, (short)0, acc[nt], false, false);
    }
    __syncthreads();               // all reads of slab ks done -> buffer free

    if (ks < 30) { stageT(ks + 2, tcur); stageA(ks + 2, acur); }
  }

  // ------------------ epilogue: row reductions ------------------
  if (tid < 64) { v2r[tid] = 0.f; hvr[tid] = 0.f; }
  if (tid == 0) *sfl = 0;
  __syncthreads();

  float hreg[64];
  #pragma unroll
  for (int j = 0; j < 8; ++j) {           // C layout: m = j + 8*hh, n = mrow
    int rl   = msub * 16 + 8 * hh + j;
    int grow = row0 + rl;
    float pv2 = 0.f, phv = 0.f;
    #pragma unroll
    for (int nt = 0; nt < 8; ++nt) {
      int n = ngrp * 128 + nt * 16 + mrow;
      float hval = h[(size_t)grow * DIM + n];
      hreg[j * 8 + nt] = hval;
      float vv = acc[nt][j];
      pv2 += vv * vv;
      phv += hval * vv;
    }
    atomicAdd(&v2r[rl], pv2);             // ds_add_f32
    atomicAdd(&hvr[rl], phv);
  }
  __syncthreads();

  // per-row scalars: z = P*h + Qv*v  (geodesic closed over span{h, v})
  if (tid < 64) {
    float h2 = h2arr[row0 + tid];
    float v2 = v2r[tid];
    float hv = hvr[tid];
    float nv = sqrtf(v2 + FEPS);
    float sw = tanhf(nv) / nv;            // exp_map: w = sw * v
    float w2 = sw * sw * v2;
    float xy = sw * hv;
    // m1 = mobius_add(-h, w) = a1*h + b1*w
    float den1 = 1.f - 2.f * xy + h2 * w2 + FEPS;
    float a1 = -(1.f - 2.f * xy + w2) / den1;
    float b1 = (1.f - h2) / den1;
    float q  = a1 * a1 * h2 + 2.f * a1 * b1 * xy + b1 * b1 * w2;
    float n1 = sqrtf(q + FEPS);
    float g  = tanhf(TPAR * atanh_clamped(n1)) / n1;   // mobius_scalar scale
    float A2 = g * a1, B2 = g * b1;                    // ms = A2*h + B2*w
    float xy2 = A2 * h2 + B2 * xy;
    float y2b = g * g * q;
    float den2 = 1.f + 2.f * xy2 + h2 * y2b + FEPS;
    float P  = (1.f + 2.f * xy2 + y2b + (1.f - h2) * A2) / den2;
    float Qv = (1.f - h2) * B2 / den2 * sw;
    Pr[tid] = P; Qr[tid] = Qv;
    if (__builtin_isnan(P) || __builtin_isnan(Qv)) atomicOr(sfl, 1);
  }
  __syncthreads();

  int nanloc = 0;
  #pragma unroll
  for (int j = 0; j < 8; ++j) {
    int rl   = msub * 16 + 8 * hh + j;
    int grow = row0 + rl;
    float P = Pr[rl], Qv = Qr[rl];
    #pragma unroll
    for (int nt = 0; nt < 8; ++nt) {
      int n = ngrp * 128 + nt * 16 + mrow;
      float z = P * hreg[j * 8 + nt] + Qv * acc[nt][j];
      nanloc |= __builtin_isnan(z) ? 1 : 0;
      out[(size_t)grow * DIM + n] = z;
    }
  }
  if (nanloc) atomicOr(sfl, 1);
  __syncthreads();
  if (tid == 0 && *sfl) atomicOr(gflag, 1);
}

// ---------------------------------------------------------------------------
// Pass 4: global NaN fallback  out = relu(h)  (early-out when flag clear)
// ---------------------------------------------------------------------------
__global__ __launch_bounds__(256) void k_fixup(const float* __restrict__ h,
                                               float* __restrict__ out,
                                               const int* __restrict__ gflag) {
  if (*gflag == 0) return;
  size_t i = ((size_t)blockIdx.x * 256 + threadIdx.x) * 4;
  float4 v = *(const float4*)(h + i);
  v.x = fmaxf(v.x, 0.f); v.y = fmaxf(v.y, 0.f);
  v.z = fmaxf(v.z, 0.f); v.w = fmaxf(v.w, 0.f);
  *(float4*)(out + i) = v;
}

extern "C" void kernel_launch(void* const* d_in, const int* in_sizes, int n_in,
                              void* d_out, int out_size, void* d_ws, size_t ws_size,
                              hipStream_t stream) {
  const float* h      = (const float*)d_in[0];   // 16384 x 1024 f32
  const float* params = (const float*)d_in[1];   // 2046 f32
  float* out = (float*)d_out;

  char* ws = (char*)d_ws;
  int*   flag  = (int*)ws;                                   // 4 B
  float* s_log = (float*)(ws + 1024);                        // 16384 f32
  float* h2    = (float*)(ws + 1024 + 65536);                // 16384 f32
  unsigned short* Tpk = (unsigned short*)(ws + 1024 + 2 * 65536);  // 2 MB bf16

  (void)hipMemsetAsync(flag, 0, sizeof(int), stream);
  k_build_T<<<1024, 256, 0, stream>>>(params, Tpk);
  k_scales<<<2048, 256, 0, stream>>>(h, s_log, h2);
  k_gemm<<<256, 1024, 142592, stream>>>(h, Tpk, s_log, h2, out, flag);
  k_fixup<<<16384, 256, 0, stream>>>(h, out, flag);
}
